// MyRNNModel_33088428048832
// MI455X (gfx1250) — matrix operand order repaired and checked by reading, not compiled, not run on hardware
//
#include <hip/hip_runtime.h>
#include <hip/hip_bf16.h>

// ---------------------------------------------------------------------------
// Problem dimensions (from the reference)
// ---------------------------------------------------------------------------
#define SEQ     512
#define BATCH   64
#define IN_DIM  512
#define HID     1024
#define OUT_DIM 512
#define NEG_SLOPE 0.01f

#define RNN_BLOCKS 32   // workgroups cooperating per recurrence step

typedef __attribute__((ext_vector_type(16))) __bf16 v16bf;
typedef __attribute__((ext_vector_type(8)))  __bf16 v8bf;
typedef __attribute__((ext_vector_type(8)))  float  v8f;

// Hardware tanh (V_TANH_F32 is a CDNA5 TRANS op).
__device__ __forceinline__ float hw_tanhf(float x) {
#if __has_builtin(__builtin_amdgcn_tanhf)
    return __builtin_amdgcn_tanhf(x);
#elif __has_builtin(__builtin_amdgcn_tanh_f32)
    return __builtin_amdgcn_tanh_f32(x);
#else
    return tanhf(x);
#endif
}

// ---------------------------------------------------------------------------
// WMMA fragment loaders (wave32; A/B use contiguous K = 16*(lane>>4) + i, not the ISA table order)
//
// A (16x32 bf16, row-major, lda elements/row):
//   lanes 0-15 : row M=lane,    K = k0+0..15 in v[0..15]
//   lanes 16-31: row M=lane-16, K = k0+16..31
//   -> two contiguous 16B loads per lane.
//
// B (32x16 bf16): per lane = one column of B, 16 consecutive K values.
//   Weights are stored TRANSPOSED (Wt[N][K] row-major) so each lane reads
//   32 contiguous bytes:
//   lanes 0-15 : col N=lane,    K = k0+0..15
//   lanes 16-31: col N=lane-16, K = k0+16..31
//
// C/D (16x16 f32, 8 VGPRs): d[v] = element (row0 + (lane>>4)*8 + v, lane&15)
// ---------------------------------------------------------------------------
__device__ __forceinline__ v16bf load_a_frag(const __bf16* __restrict__ A,
                                             int lda, int row0, int k0, int lane) {
    const int m  = lane & 15;
    const int ks = (lane >> 4) << 4;          // 0 or 16
    const __bf16* p = A + (size_t)(row0 + m) * lda + (k0 + ks);
    v8bf lo = *(const v8bf*)(p);              // K = k0+ks   .. +7
    v8bf hi = *(const v8bf*)(p + 8);          // K = k0+ks+8 .. +7
    v16bf r;
#pragma unroll
    for (int i = 0; i < 8; ++i) { r[i] = lo[i]; r[8 + i] = hi[i]; }
    return r;
}

__device__ __forceinline__ v16bf load_b_frag(const __bf16* __restrict__ Bt,
                                             int ldb, int col0, int k0, int lane) {
    const int n  = lane & 15;
    const int ko = (lane >> 4) << 4;          // 0 or 16
    return *(const v16bf*)(Bt + (size_t)(col0 + n) * ldb + (k0 + ko));
}

// B fragment from an LDS panel laid out [16 cols][HID K-values] for one n-tile.
__device__ __forceinline__ v16bf load_b_frag_lds(const __bf16* P, int k0, int lane) {
    const int n  = lane & 15;
    const int ko = (lane >> 4) << 4;          // 0 or 16
    return *(const v16bf*)(P + (size_t)n * HID + k0 + ko);
}

// ---------------------------------------------------------------------------
// Generic bf16 WMMA GEMM:  C[MxN] = act( A[MxK] * Bt[NxK]^T + bias[N] )
// One wave computes a 16x64 strip (4 n-tiles sharing one A fragment).
// ACT: 0 = identity, 1 = leaky-relu
// ---------------------------------------------------------------------------
template <int ACT>
__global__ __launch_bounds__(256) void gemm_bf16_wmma(
    const __bf16* __restrict__ A,
    const __bf16* __restrict__ Bt,
    const float*  __restrict__ bias,
    float*        __restrict__ C,
    int M, int N, int K) {
    const int wid  = (int)((blockIdx.x * blockDim.x + threadIdx.x) >> 5);
    const int lane = (int)(threadIdx.x & 31);
    const int wpr  = N >> 6;                  // waves per tile-row (4 tiles each)
    const int mt   = wid / wpr;
    const int nt0  = (wid % wpr) * 4;
    if (mt * 16 >= M) return;                 // wave-uniform guard (EXEC stays full)

    v8f acc[4];
#pragma unroll
    for (int j = 0; j < 4; ++j) {
        const float bj = bias[nt0 * 16 + j * 16 + (lane & 15)];
#pragma unroll
        for (int i = 0; i < 8; ++i) acc[j][i] = bj;
    }

    for (int k0 = 0; k0 < K; k0 += 32) {
        v16bf a = load_a_frag(A, K, mt * 16, k0, lane);
#pragma unroll
        for (int j = 0; j < 4; ++j) {
            v16bf b = load_b_frag(Bt, K, (nt0 + j) * 16, k0, lane);
            acc[j] = __builtin_amdgcn_wmma_f32_16x16x32_bf16(false, a, false, b,
                                                             (short)0, acc[j],
                                                             false, false);
        }
    }

    const int row0 = mt * 16 + ((lane >> 4) << 3);
    const int col0 = nt0 * 16 + (lane & 15);
#pragma unroll
    for (int v = 0; v < 8; ++v) {
        float* crow = C + (size_t)(row0 + v) * N + col0;
#pragma unroll
        for (int j = 0; j < 4; ++j) {
            float o = acc[j][v];
            if (ACT == 1) o = (o >= 0.f) ? o : NEG_SLOPE * o;
            crow[j * 16] = o;
        }
    }
}

// ---------------------------------------------------------------------------
// Sequential recurrence: for t in [0,512): h = tanh(xi[t] + h @ h_h)
//
// 32 blocks x 256 threads. Block b owns n-tiles {2b, 2b+1} for all 4 m-tiles
// (8 waves <-> 8 tiles). The block's 64KB B-panel (32 contiguous rows of the
// transposed h_h) is copied to LDS once; every step's B fragments then come
// from ds_load instead of L2. Device-scope barrier between steps.
// ---------------------------------------------------------------------------
__global__ __launch_bounds__(256) void rnn_recur(
    const float*  __restrict__ xi,        // [SEQ*BATCH, HID] f32
    const __bf16* __restrict__ Wt,        // [HID, HID]  (h_h transposed) bf16
    __bf16*       __restrict__ h_all,     // [(SEQ+1), BATCH, HID] bf16
    float*        __restrict__ h_final,   // [BATCH*HID] f32 (tail of d_out)
    unsigned*     __restrict__ counters)  // [SEQ], pre-zeroed
{
    __shared__ alignas(64) __bf16 bpan[2 * 16 * HID];   // 64 KB of 320 KB LDS

    const int wave = (int)(threadIdx.x >> 5);
    const int lane = (int)(threadIdx.x & 31);
    const int mt   = wave >> 1;                         // 0..3   (BATCH/16)
    const int nt   = (int)blockIdx.x * 2 + (wave & 1);  // 0..63  (HID/16)
    const int row0 = mt * 16 + ((lane >> 4) << 3);
    const int col  = nt * 16 + (lane & 15);

    // ---- one-time: stage this block's 64KB B-panel into LDS ----
    {
        const uint4* src = (const uint4*)(Wt + (size_t)blockIdx.x * 32 * HID);
        uint4*       dst = (uint4*)bpan;
        const int total = (2 * 16 * HID * 2) / 16;      // 4096 x 16B
#pragma unroll 4
        for (int i = threadIdx.x; i < total; i += 256)
            dst[i] = src[i];
        __syncthreads();
    }
    const __bf16* P = bpan + (size_t)(wave & 1) * 16 * HID;

    for (int t = 0; t < SEQ; ++t) {
        const __bf16* A  = h_all + (size_t)t * (BATCH * HID);
        const float*  xt = xi    + (size_t)t * (BATCH * HID);

        v8f acc;
#pragma unroll
        for (int v = 0; v < 8; ++v)
            acc[v] = xt[(size_t)(row0 + v) * HID + col];

#pragma unroll 4
        for (int k0 = 0; k0 < HID; k0 += 32) {
            v16bf a = load_a_frag(A, HID, mt * 16, k0, lane);
            v16bf b = load_b_frag_lds(P, k0, lane);
            acc = __builtin_amdgcn_wmma_f32_16x16x32_bf16(false, a, false, b,
                                                          (short)0, acc, false, false);
        }

        __bf16* Hn = h_all + (size_t)(t + 1) * (BATCH * HID);
        float hv[8];
#pragma unroll
        for (int v = 0; v < 8; ++v) hv[v] = hw_tanhf(acc[v]);
#pragma unroll
        for (int v = 0; v < 8; ++v)
            Hn[(size_t)(row0 + v) * HID + col] = (__bf16)hv[v];
        if (t == SEQ - 1) {                   // uniform scalar branch
#pragma unroll
            for (int v = 0; v < 8; ++v)
                h_final[(size_t)(row0 + v) * HID + col] = hv[v];
        }

        // ---- inter-block barrier (release h stores / acquire others') ----
        __threadfence();
        __syncthreads();
        if (threadIdx.x == 0) {
            __hip_atomic_fetch_add(&counters[t], 1u, __ATOMIC_ACQ_REL,
                                   __HIP_MEMORY_SCOPE_AGENT);
            while (__hip_atomic_load(&counters[t], __ATOMIC_ACQUIRE,
                                     __HIP_MEMORY_SCOPE_AGENT) < (unsigned)RNN_BLOCKS) {
                __builtin_amdgcn_s_sleep(1);
            }
        }
        __syncthreads();
        __threadfence();
    }
}

// ---------------------------------------------------------------------------
// Helpers: f32 -> bf16 cast, transposed cast, counter init
// ---------------------------------------------------------------------------
__global__ void cast_f32_bf16(const float* __restrict__ in,
                              __bf16* __restrict__ out, size_t n) {
    size_t stride = (size_t)gridDim.x * blockDim.x;
    for (size_t i = (size_t)blockIdx.x * blockDim.x + threadIdx.x; i < n; i += stride)
        out[i] = (__bf16)in[i];
}

// W: [K x N] f32 row-major  ->  Wt: [N x K] bf16 row-major
__global__ void transpose_cast(const float* __restrict__ W,
                               __bf16* __restrict__ Wt, int K, int N) {
    size_t n = (size_t)K * N;
    size_t stride = (size_t)gridDim.x * blockDim.x;
    for (size_t i = (size_t)blockIdx.x * blockDim.x + threadIdx.x; i < n; i += stride) {
        int k = (int)(i / N);
        int c = (int)(i % N);
        Wt[(size_t)c * K + k] = (__bf16)W[i];
    }
}

__global__ void init_counters(unsigned* c, int n) {
    int i = blockIdx.x * blockDim.x + threadIdx.x;
    if (i < n) c[i] = 0;
}

// ---------------------------------------------------------------------------
// Launcher
// ---------------------------------------------------------------------------
extern "C" void kernel_launch(void* const* d_in, const int* in_sizes, int n_in,
                              void* d_out, int out_size, void* d_ws, size_t ws_size,
                              hipStream_t stream) {
    const float* x   = (const float*)d_in[0];  // [SEQ, BATCH, IN_DIM]
    const float* h0  = (const float*)d_in[1];  // [BATCH, HID]
    const float* i_h = (const float*)d_in[2];  // [IN_DIM, HID]
    const float* h_h = (const float*)d_in[3];  // [HID, HID]
    const float* h_o = (const float*)d_in[4];  // [HID, OUT_DIM]
    const float* b_i = (const float*)d_in[5];  // [HID]
    const float* b_o = (const float*)d_in[6];  // [OUT_DIM]

    float* out       = (float*)d_out;                                   // [SEQ*BATCH*OUT_DIM]
    float* out_final = out + (size_t)SEQ * BATCH * OUT_DIM;             // [BATCH*HID]

    // ---- workspace carve-out (all section sizes are multiples of 256B) ----
    char* w = (char*)d_ws;
    float*  xi    = (float*)w;  w += (size_t)SEQ * BATCH * HID * sizeof(float);
    __bf16* h_all = (__bf16*)w; w += (size_t)(SEQ + 1) * BATCH * HID * sizeof(__bf16);
    __bf16* x_bf  = (__bf16*)w; w += (size_t)SEQ * BATCH * IN_DIM * sizeof(__bf16);
    __bf16* iht   = (__bf16*)w; w += (size_t)HID * IN_DIM * sizeof(__bf16);
    __bf16* hht   = (__bf16*)w; w += (size_t)HID * HID * sizeof(__bf16);
    __bf16* hot   = (__bf16*)w; w += (size_t)OUT_DIM * HID * sizeof(__bf16);
    unsigned* counters = (unsigned*)w;

    // ---- phase 0: precision conversion + weight transposes ----
    cast_f32_bf16<<<2048, 256, 0, stream>>>(x, x_bf, (size_t)SEQ * BATCH * IN_DIM);
    cast_f32_bf16<<<(BATCH * HID + 255) / 256, 256, 0, stream>>>(h0, h_all,
                                                                 (size_t)BATCH * HID);
    transpose_cast<<<1024, 256, 0, stream>>>(i_h, iht, IN_DIM, HID);
    transpose_cast<<<1024, 256, 0, stream>>>(h_h, hht, HID, HID);
    transpose_cast<<<1024, 256, 0, stream>>>(h_o, hot, HID, OUT_DIM);
    init_counters<<<(SEQ + 255) / 256, 256, 0, stream>>>(counters, SEQ);

    // ---- phase 1: xi = x @ i_h + b_i   (M=32768, N=1024, K=512) ----
    {
        const int M = SEQ * BATCH, N = HID, K = IN_DIM;
        const int waves = (M / 16) * (N / 64);     // 32768
        gemm_bf16_wmma<0><<<waves / 8, 256, 0, stream>>>(x_bf, iht, b_i, xi, M, N, K);
    }

    // ---- phase 2: sequential tanh recurrence across 32 cooperating blocks ----
    rnn_recur<<<RNN_BLOCKS, 256, 0, stream>>>(xi, hht, h_all, out_final, counters);

    // ---- phase 3: out = leaky(h_t @ h_o + b_o)  (M=32768, N=512, K=1024) ----
    {
        const int M = SEQ * BATCH, N = OUT_DIM, K = HID;
        const int waves = (M / 16) * (N / 64);     // 16384
        gemm_bf16_wmma<1><<<waves / 8, 256, 0, stream>>>(h_all + (size_t)BATCH * HID,
                                                         hot, b_o, out, M, N, K);
    }
}
